// MDCN_32839319945601
// MI455X (gfx1250) — matrix operand order repaired and checked by reading, not compiled, not run on hardware
//
#include <hip/hip_runtime.h>
#include <hip/hip_bf16.h>
#include <math.h>

// ---------------- types for WMMA ----------------
typedef __attribute__((ext_vector_type(16))) __bf16 v16bf;
typedef __attribute__((ext_vector_type(8)))  float  v8f;

union FragU { uint4 q[2]; v16bf v; };

__device__ __forceinline__ unsigned short f2bf(float f) {
    unsigned int u = __float_as_uint(f);
    u += 0x7FFFu + ((u >> 16) & 1u);          // round-to-nearest-even
    return (unsigned short)(u >> 16);
}

// stats float layout inside ws:
// [2]=scale0 [3]=shift0 ; [76..111]=scaleCh ; [112..147]=shiftCh ;
// [148..347]=scale2 ; [348..547]=shift2
#define SC0   2
#define SH0   3
#define SCCH  76
#define SHCH  112
#define SC2   148
#define SH2   348

// ---------------- 1) gather + BN0 partials ----------------
__global__ void k_gather_x(const float* __restrict__ ent, const float* __restrict__ rel,
                           const int* __restrict__ eid, const int* __restrict__ rid,
                           float* __restrict__ x, float* __restrict__ part0, int B) {
    int idx = blockIdx.x * 256 + threadIdx.x;
    float v = 0.f;
    if (idx < B * 400) {
        int b = idx / 400, k = idx % 400, col = k >> 1;
        v = (k & 1) ? rel[(size_t)rid[b] * 200 + col] : ent[(size_t)eid[b] * 200 + col];
        x[idx] = v;
    }
    __shared__ float ss[256], sq[256];
    ss[threadIdx.x] = v; sq[threadIdx.x] = v * v;
    __syncthreads();
    for (int s = 128; s > 0; s >>= 1) {
        if (threadIdx.x < s) { ss[threadIdx.x] += ss[threadIdx.x + s]; sq[threadIdx.x] += sq[threadIdx.x + s]; }
        __syncthreads();
    }
    if (threadIdx.x == 0) { part0[blockIdx.x * 2] = ss[0]; part0[blockIdx.x * 2 + 1] = sq[0]; }
}

__global__ void k_fin0(const float* __restrict__ part0, int nPart,
                       float* __restrict__ stats,
                       const float* __restrict__ g0, const float* __restrict__ b0, int B) {
    __shared__ float ss[256], sq[256];
    float s = 0.f, q = 0.f;
    for (int i = threadIdx.x; i < nPart; i += 256) { s += part0[i * 2]; q += part0[i * 2 + 1]; }
    ss[threadIdx.x] = s; sq[threadIdx.x] = q;
    __syncthreads();
    for (int st = 128; st > 0; st >>= 1) {
        if (threadIdx.x < st) { ss[threadIdx.x] += ss[threadIdx.x + st]; sq[threadIdx.x] += sq[threadIdx.x + st]; }
        __syncthreads();
    }
    if (threadIdx.x == 0) {
        float n = (float)B * 400.f;
        float m = ss[0] / n, var = sq[0] / n - m * m;
        float sc = g0[0] * rsqrtf(var + 1e-5f);
        stats[SC0] = sc; stats[SH0] = b0[0] - m * sc;
    }
}

// ---------------- per-sample dynamic conv ----------------
__device__ __forceinline__ float conv_val(int c, int p,
                                          const float* xs, const float* f1s,
                                          const float* f3s, const float* f5s) {
    int i = p / 20, j = p % 20;
    float acc = 0.f;
    if (c < 8) {                              // 1x5, pad W=2
        #pragma unroll
        for (int dj = 0; dj < 5; ++dj) {
            int jj = j + dj - 2;
            if (jj >= 0 && jj < 20) acc += xs[i * 20 + jj] * f1s[c * 5 + dj];
        }
    } else if (c < 28) {                      // 3x3, pad 1
        int cc = c - 8;
        #pragma unroll
        for (int di = 0; di < 3; ++di) {
            int ii = i + di - 1;
            if (ii < 0 || ii >= 20) continue;
            #pragma unroll
            for (int dj = 0; dj < 3; ++dj) {
                int jj = j + dj - 1;
                if (jj >= 0 && jj < 20) acc += xs[ii * 20 + jj] * f3s[cc * 9 + di * 3 + dj];
            }
        }
    } else {                                  // 1x9, pad W=4
        int cc = c - 28;
        #pragma unroll
        for (int dj = 0; dj < 9; ++dj) {
            int jj = j + dj - 4;
            if (jj >= 0 && jj < 20) acc += xs[i * 20 + jj] * f5s[cc * 9 + dj];
        }
    }
    return acc;
}

__device__ __forceinline__ void conv_load_smem(int b, int tid,
                                               const float* __restrict__ x, const int* __restrict__ rid,
                                               const float* __restrict__ filt1, const float* __restrict__ filt3,
                                               const float* __restrict__ filt5, const float* __restrict__ stats,
                                               float* xs, float* f1s, float* f3s, float* f5s) {
    float sc0 = stats[SC0], sh0 = stats[SH0];
    int r = rid[b];
    for (int i = tid; i < 400; i += 256) xs[i] = x[(size_t)b * 400 + i] * sc0 + sh0;
    for (int i = tid; i < 40;  i += 256) f1s[i] = filt1[(size_t)r * 40 + i];
    for (int i = tid; i < 180; i += 256) f3s[i] = filt3[(size_t)r * 180 + i];
    for (int i = tid; i < 72;  i += 256) f5s[i] = filt5[(size_t)r * 72 + i];
}

// 2) conv stats pass: per-block per-channel partial sums (deterministic tree reduce)
__global__ void k_conv_stats(const float* __restrict__ x, const int* __restrict__ rid,
                             const float* __restrict__ filt1, const float* __restrict__ filt3,
                             const float* __restrict__ filt5, const float* __restrict__ stats,
                             float* __restrict__ chanPart) {
    __shared__ float xs[400], f1s[40], f3s[180], f5s[72];
    __shared__ float red[256], redq[256];
    int b = blockIdx.x, tid = threadIdx.x;
    conv_load_smem(b, tid, x, rid, filt1, filt3, filt5, stats, xs, f1s, f3s, f5s);
    __syncthreads();
    for (int c = 0; c < 36; ++c) {
        float s = 0.f, q = 0.f;
        for (int p = tid; p < 400; p += 256) {
            float v = conv_val(c, p, xs, f1s, f3s, f5s);
            s += v; q += v * v;
        }
        red[tid] = s; redq[tid] = q;
        __syncthreads();
        for (int st = 128; st > 0; st >>= 1) {
            if (tid < st) { red[tid] += red[tid + st]; redq[tid] += redq[tid + st]; }
            __syncthreads();
        }
        if (tid == 0) { chanPart[(size_t)b * 72 + c] = red[0]; chanPart[(size_t)b * 72 + 36 + c] = redq[0]; }
        __syncthreads();
    }
}

__global__ void k_finch(const float* __restrict__ chanPart, int B, float* __restrict__ stats,
                        const float* __restrict__ g11, const float* __restrict__ b11,
                        const float* __restrict__ g12, const float* __restrict__ b12,
                        const float* __restrict__ g13, const float* __restrict__ b13) {
    int c = blockIdx.x;                       // 0..35
    __shared__ float ss[256], sq[256];
    float s = 0.f, q = 0.f;
    for (int b = threadIdx.x; b < B; b += 256) {
        s += chanPart[(size_t)b * 72 + c];
        q += chanPart[(size_t)b * 72 + 36 + c];
    }
    ss[threadIdx.x] = s; sq[threadIdx.x] = q;
    __syncthreads();
    for (int st = 128; st > 0; st >>= 1) {
        if (threadIdx.x < st) { ss[threadIdx.x] += ss[threadIdx.x + st]; sq[threadIdx.x] += sq[threadIdx.x + st]; }
        __syncthreads();
    }
    if (threadIdx.x == 0) {
        float n = (float)B * 400.f;
        float m = ss[0] / n, var = sq[0] / n - m * m;
        float g, bb;
        if (c < 8)       { g = g11[c];      bb = b11[c]; }
        else if (c < 28) { g = g12[c - 8];  bb = b12[c - 8]; }
        else             { g = g13[c - 28]; bb = b13[c - 28]; }
        float sc = g * rsqrtf(var + 1e-5f);
        stats[SCCH + c] = sc; stats[SHCH + c] = bb - m * sc;
    }
}

// 3) conv apply: BN + ReLU -> bf16 feature matrix [B, 14400]
__global__ void k_conv_apply(const float* __restrict__ x, const int* __restrict__ rid,
                             const float* __restrict__ filt1, const float* __restrict__ filt3,
                             const float* __restrict__ filt5, const float* __restrict__ stats,
                             unsigned short* __restrict__ feat) {
    __shared__ float xs[400], f1s[40], f3s[180], f5s[72];
    int b = blockIdx.x, tid = threadIdx.x;
    conv_load_smem(b, tid, x, rid, filt1, filt3, filt5, stats, xs, f1s, f3s, f5s);
    __syncthreads();
    for (int idx = tid; idx < 14400; idx += 256) {
        int c = idx / 400, p = idx % 400;
        float v = conv_val(c, p, xs, f1s, f3s, f5s) * stats[SCCH + c] + stats[SHCH + c];
        v = v > 0.f ? v : 0.f;
        feat[(size_t)b * 14400 + idx] = f2bf(v);
    }
}

// ---------------- 4) one-time bf16 operand converts ----------------
__global__ void k_fcwT(const float* __restrict__ fc_w, unsigned short* __restrict__ fcwT) {
    int idx = blockIdx.x * 256 + threadIdx.x;
    if (idx >= 208 * 14400) return;
    int n = idx / 14400, k = idx % 14400;
    float v = (n < 200) ? fc_w[(size_t)k * 200 + n] : 0.f;
    fcwT[idx] = f2bf(v);
}

__global__ void k_entcvt(const float* __restrict__ ent, unsigned short* __restrict__ entbf) {
    int idx = blockIdx.x * 256 + threadIdx.x;
    if (idx >= 40000 * 224) return;
    int n = idx / 224, k = idx % 224;
    float v = (k < 200) ? ent[(size_t)n * 200 + k] : 0.f;
    entbf[idx] = f2bf(v);
}

// ---------------- WMMA fragment loads ----------------
// A lane(l): row m0+(l&15); elems e: K = kk + e + 8*(l>>4) + (e>=8 ? 8 : 0)
// B lane(l): col n0+(l&15) (row of transposed operand); elems e: K = kk + e + 8*(l>>4) + (e>=8 ? 8 : 0)
__device__ __forceinline__ v16bf load_afrag(const unsigned short* __restrict__ arow, int kk, int hh) {
    FragU a;
    const uint4* ap = (const uint4*)(arow + kk + 8 * hh);
    a.q[0] = ap[0]; a.q[1] = ap[2];           // +32 bytes = elems 8..15
    return a.v;
}
__device__ __forceinline__ v16bf load_bfrag(const unsigned short* __restrict__ brow, int kk, int hh) {
    FragU b;
    const uint4* bp = (const uint4*)(brow + kk + 8 * hh);
    b.q[0] = bp[0]; b.q[1] = bp[2];
    return b.v;
}

// 5) GEMM1: feat[B,14400](bf16) x fc_wT[208,14400](bf16) -> h_raw[B,200] (f32) + fc_b
__global__ void k_gemm1(const unsigned short* __restrict__ feat,
                        const unsigned short* __restrict__ fcwT,
                        const float* __restrict__ fc_b,
                        float* __restrict__ h_raw, int B) {
    int lane = threadIdx.x & 31, wave = threadIdx.x >> 5;
    int n_tile = blockIdx.x * 8 + wave;
    if (n_tile >= 13) return;                 // wave-uniform
    int m0 = blockIdx.y * 16, n0 = n_tile * 16;
    int hh = lane >> 4, l15 = lane & 15;
    const unsigned short* arow = feat + (size_t)(m0 + l15) * 14400;
    const unsigned short* brow = fcwT + (size_t)(n0 + l15) * 14400;
    v8f acc = {};
    for (int kk = 0; kk < 14400; kk += 32) {
        v16bf a = load_afrag(arow, kk, hh);
        v16bf b = load_bfrag(brow, kk, hh);
        acc = __builtin_amdgcn_wmma_f32_16x16x32_bf16(
            false, a, false, b, (short)0, acc, false, false);
    }
    int n = n0 + l15;
    if (n < 200) {
        float fb = fc_b[n];
        #pragma unroll
        for (int r = 0; r < 8; ++r) {
            int m = m0 + 8 * hh + r;
            if (m < B) h_raw[(size_t)m * 200 + n] = acc[r] + fb;
        }
    }
}

// BN1d stats per column (deterministic) + finalize scale/shift
__global__ void k_colstats(const float* __restrict__ h_raw, int B, float* __restrict__ stats,
                           const float* __restrict__ g2, const float* __restrict__ b2) {
    int c = blockIdx.x;                       // 0..199
    __shared__ float ss[256], sq[256];
    float s = 0.f, q = 0.f;
    for (int r = threadIdx.x; r < B; r += 256) {
        float v = h_raw[(size_t)r * 200 + c];
        s += v; q += v * v;
    }
    ss[threadIdx.x] = s; sq[threadIdx.x] = q;
    __syncthreads();
    for (int st = 128; st > 0; st >>= 1) {
        if (threadIdx.x < st) { ss[threadIdx.x] += ss[threadIdx.x + st]; sq[threadIdx.x] += sq[threadIdx.x + st]; }
        __syncthreads();
    }
    if (threadIdx.x == 0) {
        float n = (float)B;
        float m = ss[0] / n, var = sq[0] / n - m * m;
        float sc = g2[c] * rsqrtf(var + 1e-5f);
        stats[SC2 + c] = sc; stats[SH2 + c] = b2[c] - m * sc;
    }
}

// BN1d + ReLU -> padded bf16 A matrix [B, 224]
__global__ void k_abf(const float* __restrict__ h_raw, const float* __restrict__ stats,
                      unsigned short* __restrict__ Abf, int B) {
    int idx = blockIdx.x * 256 + threadIdx.x;
    if (idx >= B * 224) return;
    int b = idx / 224, k = idx % 224;
    float v = 0.f;
    if (k < 200) {
        v = h_raw[(size_t)b * 200 + k] * stats[SC2 + k] + stats[SH2 + k];
        v = v > 0.f ? v : 0.f;
    }
    Abf[idx] = f2bf(v);
}

// 6) GEMM2: Abf[B,224] x entbf[40000,224]^T + bias -> sigmoid -> out[B,40000] f32
// Each wave: 1 M-tile x 4 N-tiles (A fragment loaded once per k-step, 4 WMMAs issued on it).
__global__ void k_gemm2(const unsigned short* __restrict__ Abf,
                        const unsigned short* __restrict__ entbf,
                        const float* __restrict__ bias,
                        float* __restrict__ out, int B) {
    int lane = threadIdx.x & 31, wave = threadIdx.x >> 5;
    int n_group = blockIdx.x * 8 + wave;      // 2500/4 = 625 groups of 4 n-tiles
    if (n_group >= 625) return;               // wave-uniform
    int m0 = blockIdx.y * 16, n0 = n_group * 64;
    int hh = lane >> 4, l15 = lane & 15;
    const unsigned short* arow = Abf + (size_t)(m0 + l15) * 224;
    const unsigned short* brow = entbf + (size_t)(n0 + l15) * 224;
    v8f acc0 = {}, acc1 = {}, acc2 = {}, acc3 = {};
    #pragma unroll
    for (int kk = 0; kk < 224; kk += 32) {
        v16bf a  = load_afrag(arow, kk, hh);
        v16bf b0 = load_bfrag(brow,              kk, hh);
        v16bf b1 = load_bfrag(brow + 16 * 224,   kk, hh);
        v16bf b2 = load_bfrag(brow + 32 * 224,   kk, hh);
        v16bf b3 = load_bfrag(brow + 48 * 224,   kk, hh);
        acc0 = __builtin_amdgcn_wmma_f32_16x16x32_bf16(false, a, false, b0, (short)0, acc0, false, false);
        acc1 = __builtin_amdgcn_wmma_f32_16x16x32_bf16(false, a, false, b1, (short)0, acc1, false, false);
        acc2 = __builtin_amdgcn_wmma_f32_16x16x32_bf16(false, a, false, b2, (short)0, acc2, false, false);
        acc3 = __builtin_amdgcn_wmma_f32_16x16x32_bf16(false, a, false, b3, (short)0, acc3, false, false);
    }
    // epilogue: bias + sigmoid, 4 sub-tiles
    #pragma unroll
    for (int sub = 0; sub < 4; ++sub) {
        v8f acc = (sub == 0) ? acc0 : (sub == 1) ? acc1 : (sub == 2) ? acc2 : acc3;
        int n = n0 + sub * 16 + l15;
        float bn = bias[n];
        #pragma unroll
        for (int r = 0; r < 8; ++r) {
            int m = m0 + 8 * hh + r;
            if (m < B) {
                float v = acc[r] + bn;
                out[(size_t)m * 40000 + n] = 1.f / (1.f + __expf(-v));
            }
        }
    }
}

// ---------------- host ----------------
static inline size_t alignup(size_t x) { return (x + 255) & ~(size_t)255; }

extern "C" void kernel_launch(void* const* d_in, const int* in_sizes, int n_in,
                              void* d_out, int out_size, void* d_ws, size_t ws_size,
                              hipStream_t stream) {
    const float* ent_emb = (const float*)d_in[0];
    const float* rel_emb = (const float*)d_in[1];
    const float* filt1   = (const float*)d_in[2];
    const float* filt3   = (const float*)d_in[3];
    const float* filt5   = (const float*)d_in[4];
    const float* fc_w    = (const float*)d_in[5];
    const float* fc_b    = (const float*)d_in[6];
    const float* bias    = (const float*)d_in[7];
    const float* g0  = (const float*)d_in[8];
    const float* b0  = (const float*)d_in[9];
    const float* g11 = (const float*)d_in[10];
    const float* b11 = (const float*)d_in[11];
    const float* g12 = (const float*)d_in[12];
    const float* b12 = (const float*)d_in[13];
    const float* g13 = (const float*)d_in[14];
    const float* b13 = (const float*)d_in[15];
    const float* g2  = (const float*)d_in[16];
    const float* b2  = (const float*)d_in[17];
    const int* entity_id   = (const int*)d_in[18];
    const int* relation_id = (const int*)d_in[19];
    float* out = (float*)d_out;

    const int B = in_sizes[18];               // 2048, multiple of 16

    // workspace partition
    char* ws = (char*)d_ws;
    size_t off = 0;
    float* stats = (float*)(ws + off);                 off = alignup(off + 548 * 4);
    float* part0 = (float*)(ws + off);
    int nPart0 = (B * 400 + 255) / 256;                off = alignup(off + (size_t)nPart0 * 2 * 4);
    float* xbuf  = (float*)(ws + off);                 off = alignup(off + (size_t)B * 400 * 4);
    float* chanPart = (float*)(ws + off);              off = alignup(off + (size_t)B * 72 * 4);
    unsigned short* feat  = (unsigned short*)(ws + off); off = alignup(off + (size_t)B * 14400 * 2);
    unsigned short* fcwT  = (unsigned short*)(ws + off); off = alignup(off + (size_t)208 * 14400 * 2);
    unsigned short* entbf = (unsigned short*)(ws + off); off = alignup(off + (size_t)40000 * 224 * 2);
    float* h_raw = (float*)(ws + off);                 off = alignup(off + (size_t)B * 200 * 4);
    unsigned short* Abf = (unsigned short*)(ws + off); off = alignup(off + (size_t)B * 224 * 2);

    // independent operand converts
    k_fcwT<<<(208 * 14400 + 255) / 256, 256, 0, stream>>>(fc_w, fcwT);
    k_entcvt<<<(40000 * 224 + 255) / 256, 256, 0, stream>>>(ent_emb, entbf);

    // 1) gather + BN0
    k_gather_x<<<nPart0, 256, 0, stream>>>(ent_emb, rel_emb, entity_id, relation_id, xbuf, part0, B);
    k_fin0<<<1, 256, 0, stream>>>(part0, nPart0, stats, g0, b0, B);

    // 2-3) dynamic convs: stats pass, finalize, apply pass
    k_conv_stats<<<B, 256, 0, stream>>>(xbuf, relation_id, filt1, filt3, filt5, stats, chanPart);
    k_finch<<<36, 256, 0, stream>>>(chanPart, B, stats, g11, b11, g12, b12, g13, b13);
    k_conv_apply<<<B, 256, 0, stream>>>(xbuf, relation_id, filt1, filt3, filt5, stats, feat);

    // 5) FC GEMM (WMMA) + BN1d
    k_gemm1<<<dim3(2, B / 16), 256, 0, stream>>>(feat, fcwT, fc_b, h_raw, B);
    k_colstats<<<200, 256, 0, stream>>>(h_raw, B, stats, g2, b2);
    k_abf<<<(B * 224 + 255) / 256, 256, 0, stream>>>(h_raw, stats, Abf, B);

    // 6) logits GEMM (WMMA) + bias + sigmoid: 625 groups of 4 N-tiles
    k_gemm2<<<dim3((625 + 7) / 8, B / 16), 256, 0, stream>>>(Abf, entbf, bias, out, B);
}
